// PointTransformerBlock_83107617177693
// MI455X (gfx1250) — matrix operand order repaired and checked by reading, not compiled, not run on hardware
//
#include <hip/hip_runtime.h>

typedef __attribute__((ext_vector_type(16))) _Float16 v16h;
typedef __attribute__((ext_vector_type(8)))  _Float16 v8h;
typedef __attribute__((ext_vector_type(4)))  _Float16 v4h;
typedef __attribute__((ext_vector_type(8)))  float    v8f;

#define R_TOTAL 16384   // B*N
#define NPTS    4096
#define KNN     16

// ---------------------------------------------------------------------------
// WMMA fragment helpers (CDNA5 wave32 layouts per CDNA5 ISA)
// A 16x32 f16: m=lane%16, g=lane/16; halves 0..7 -> K=k0+g*8+0..7,
//              halves 8..15 -> K=k0+16+g*8+0..7  (two contiguous 16B runs)
// B 32x16 f16 from transposed weights Bt[N][K]: same K order as A,
//              column n = lane%16 (two contiguous 16B runs per lane)
// ---------------------------------------------------------------------------
__device__ __forceinline__ v16h load_a_frag(const _Float16* A, int lda, int k0) {
  int lane = threadIdx.x & 31;
  int m = lane & 15, g = lane >> 4;
  const _Float16* base = A + (size_t)m * lda + k0 + g * 8;
  v8h lo = *(const v8h*)(base);
  v8h hi = *(const v8h*)(base + 16);
  v16h r;
#pragma unroll
  for (int i = 0; i < 8; ++i) { r[i] = lo[i]; r[8 + i] = hi[i]; }
  return r;
}

__device__ __forceinline__ v16h load_bt_frag(const _Float16* Bt, int ldk, int n0, int k0) {
  int lane = threadIdx.x & 31;
  int n = lane & 15, g = lane >> 4;
  const _Float16* base = Bt + (size_t)(n0 + n) * ldk + k0 + g * 8;
  v8h lo = *(const v8h*)(base);
  v8h hi = *(const v8h*)(base + 16);
  v16h r;
#pragma unroll
  for (int i = 0; i < 8; ++i) { r[i] = lo[i]; r[8 + i] = hi[i]; }
  return r;
}

__device__ __forceinline__ v8f wmma16(v16h a, v16h b, v8f c) {
  return __builtin_amdgcn_wmma_f32_16x16x32_f16(false, a, false, b, (short)0, c,
                                                false, false);
}

// ---------------------------------------------------------------------------
// Small utility kernels
// ---------------------------------------------------------------------------
__global__ void zero_f32(float* p, int n) {
  int i = blockIdx.x * blockDim.x + threadIdx.x;
  if (i < n) p[i] = 0.0f;
}

__global__ void f32_to_f16_v4(const float* __restrict__ src,
                              _Float16* __restrict__ dst, int n4) {
  int i = blockIdx.x * blockDim.x + threadIdx.x;
  if (i >= n4) return;
  float4 v = ((const float4*)src)[i];
  v4h h = { (_Float16)v.x, (_Float16)v.y, (_Float16)v.z, (_Float16)v.w };
  ((v4h*)dst)[i] = h;
}

// transpose-convert weights: src is KxN row-major f32, dst is NxK f16
__global__ void wt_prep(const float* __restrict__ src, _Float16* __restrict__ dst,
                        int K, int N) {
  int i = blockIdx.x * blockDim.x + threadIdx.x;
  if (i >= K * N) return;
  int k = i / N, n = i - k * N;
  dst[(size_t)n * K + k] = (_Float16)src[i];
}

__global__ void copy_f32_v4(const float* __restrict__ src, float* __restrict__ dst,
                            int n4) {
  int i = blockIdx.x * blockDim.x + threadIdx.x;
  if (i < n4) ((float4*)dst)[i] = ((const float4*)src)[i];
}

// ---------------------------------------------------------------------------
// Generic WMMA GEMM: out[R x N] = A_f16[R x 64] @ Bt_f16[N x 64]^T + bias
// One wave per 16-row tile; 8 waves per block. bias is never null (use a
// zeroed dummy for bias-free GEMMs to keep the epilogue branch-free).
// ---------------------------------------------------------------------------
__global__ void __launch_bounds__(256) gemm_k64(
    const _Float16* __restrict__ A, const _Float16* __restrict__ Bt,
    const float* __restrict__ bias, float* __restrict__ out, int R, int N) {
  int wave = blockIdx.x * (blockDim.x >> 5) + (threadIdx.x >> 5);
  int row0 = wave * 16;
  if (row0 >= R) return;
  int lane = threadIdx.x & 31;
  int g = lane >> 4, nn = lane & 15;

  v16h a0 = load_a_frag(A + (size_t)row0 * 64, 64, 0);
  v16h a1 = load_a_frag(A + (size_t)row0 * 64, 64, 32);

  for (int n0 = 0; n0 < N; n0 += 16) {
    v8f acc = {};
    acc = wmma16(a0, load_bt_frag(Bt, 64, n0, 0), acc);
    acc = wmma16(a1, load_bt_frag(Bt, 64, n0, 32), acc);
    int c = n0 + nn;
    float bv = bias[c];
    int rbase = row0 + 8 * g;
#pragma unroll
    for (int r = 0; r < 8; ++r)
      out[(size_t)(rbase + r) * N + c] = acc[r] + bv;
  }
}

// ---------------------------------------------------------------------------
// Channel statistics: sum / sumsq per channel (C=64) over R rows
// ---------------------------------------------------------------------------
__global__ void chan_stats(const float* __restrict__ in, float* __restrict__ sum,
                           float* __restrict__ sq, int R) {
  int t = blockIdx.x * blockDim.x + threadIdx.x;
  int c = t & 63;
  int r0 = t >> 6;
  int rs = (gridDim.x * blockDim.x) >> 6;
  float s = 0.f, q = 0.f;
  for (int r = r0; r < R; r += rs) {
    float v = in[(size_t)r * 64 + c];
    s += v;
    q += v * v;
  }
  atomicAdd(&sum[c], s);
  atomicAdd(&sq[c], q);
}

// ---------------------------------------------------------------------------
// BN apply (vectorized x4): v = relu(g*(x-mu)*rsqrt(var+eps)+b); optional
// residual added AFTER relu; optional f32 and f16 outputs.
// ---------------------------------------------------------------------------
__global__ void bn_apply_v4(const float* __restrict__ in, const float* __restrict__ sum,
                            const float* __restrict__ sq, const float* __restrict__ gm,
                            const float* __restrict__ bt, const float* __restrict__ res,
                            float* __restrict__ out32, _Float16* __restrict__ out16,
                            int R) {
  int i = blockIdx.x * blockDim.x + threadIdx.x;   // vector index
  if (i >= (R * 64) / 4) return;
  int c0 = (i * 4) & 63;
  float invR = 1.0f / (float)R;
  float4 v = ((const float4*)in)[i];
  float o[4] = { v.x, v.y, v.z, v.w };
#pragma unroll
  for (int k = 0; k < 4; ++k) {
    int c = c0 + k;
    float mu = sum[c] * invR;
    float var = sq[c] * invR - mu * mu;
    float t = gm[c] * (o[k] - mu) * rsqrtf(var + 1e-5f) + bt[c];
    o[k] = fmaxf(t, 0.0f);
  }
  if (res) {
    float4 rr = ((const float4*)res)[i];
    o[0] += rr.x; o[1] += rr.y; o[2] += rr.z; o[3] += rr.w;
  }
  if (out32) {
    float4 w = { o[0], o[1], o[2], o[3] };
    ((float4*)out32)[i] = w;
  }
  if (out16) {
    v4h h = { (_Float16)o[0], (_Float16)o[1], (_Float16)o[2], (_Float16)o[3] };
    ((v4h*)out16)[i] = h;
  }
}

// ---------------------------------------------------------------------------
// KNN: per-thread query, candidate points tiled through LDS, 16-deep
// insertion sort (ascending d2, self included, matches top_k(-d2)).
// ---------------------------------------------------------------------------
__global__ void __launch_bounds__(256) knn_kernel(const float* __restrict__ p,
                                                  int* __restrict__ idx) {
  int qid = blockIdx.x * blockDim.x + threadIdx.x;   // 0..16383
  int b = qid >> 12;
  int n = qid & (NPTS - 1);
  const float* pb = p + (size_t)b * NPTS * 3;
  float qx = pb[n * 3 + 0], qy = pb[n * 3 + 1], qz = pb[n * 3 + 2];

  float bd[KNN];
  int bi[KNN];
#pragma unroll
  for (int j = 0; j < KNN; ++j) { bd[j] = 3.0e38f; bi[j] = 0; }

  __shared__ float sp[256 * 3];
  for (int t0 = 0; t0 < NPTS; t0 += 256) {
    __syncthreads();
    for (int k = threadIdx.x; k < 256 * 3; k += 256) sp[k] = pb[(size_t)t0 * 3 + k];
    // prefetch next tile while this one is processed (global_prefetch_b8)
    if (t0 + 256 < NPTS)
      __builtin_prefetch(pb + (size_t)(t0 + 256) * 3 + threadIdx.x * 3, 0, 1);
    __syncthreads();
    for (int m = 0; m < 256; ++m) {
      float dx = qx - sp[m * 3 + 0];
      float dy = qy - sp[m * 3 + 1];
      float dz = qz - sp[m * 3 + 2];
      float d = dx * dx + dy * dy + dz * dz;
      if (d < bd[KNN - 1]) {
        int j = KNN - 1;
        while (j > 0 && bd[j - 1] > d) {
          bd[j] = bd[j - 1];
          bi[j] = bi[j - 1];
          --j;
        }
        bd[j] = d;
        bi[j] = t0 + m;
      }
    }
  }
#pragma unroll
  for (int j = 0; j < KNN; ++j) idx[(size_t)qid * KNN + j] = bi[j];
}

// ---------------------------------------------------------------------------
// Fused per-point attention kernel. One wave per point (M=16 neighbors =
// WMMA M-dim). 2 waves per block; all intermediates in LDS.
//   pe   = relu(rel_pos@pos_w1+b1)@pos_w2+b2          (4 WMMA)
//   hid  = relu((q-kj+pe)@attn_w1+b1)   16x256        (32 WMMA)
//   sim  = hid@attn_w2+b2               16x64         (32 WMMA)
//   agg  = sum_j softmax_j(sim) * (vj+pe)
// ---------------------------------------------------------------------------
#define AW 2
__global__ void __launch_bounds__(64) attn_kernel(
    const float* __restrict__ qkv, const float* __restrict__ p,
    const int* __restrict__ idx, const float* __restrict__ pos_w1,
    const float* __restrict__ pos_b1, const _Float16* __restrict__ pos_w2t,
    const float* __restrict__ pos_b2, const _Float16* __restrict__ w1t,
    const float* __restrict__ attn_b1, const _Float16* __restrict__ w2t,
    const float* __restrict__ attn_b2, float* __restrict__ agg) {
  __shared__ __align__(16) float    s_qk[AW][16][64];
  __shared__ __align__(16) float    s_vv[AW][16][64];
  __shared__ __align__(16) float    s_sim[AW][16][64];
  __shared__ __align__(16) _Float16 s_x1[AW][16][64];
  __shared__ __align__(16) _Float16 s_peh[AW][16][32];
  __shared__ __align__(16) _Float16 s_hid[AW][16][256];
  __shared__ __align__(16) float    s_q[AW][64];
  __shared__ float s_rel[AW][16][3];
  __shared__ int   s_id[AW][16];

  int w = threadIdx.x >> 5;
  int lane = threadIdx.x & 31;
  int pt = blockIdx.x * AW + w;          // 0..16383
  int b = pt >> 12;
  int n = pt & (NPTS - 1);
  int g = lane >> 4, nn = lane & 15;
  const float* pb = p + (size_t)b * NPTS * 3;
  const float* qrow = qkv + (size_t)pt * 192;

  // --- phase A: load q, neighbor indices, own position
  if (lane < 16) s_id[w][lane] = idx[(size_t)pt * KNN + lane];
  s_q[w][lane] = qrow[lane];
  s_q[w][lane + 32] = qrow[lane + 32];
  float px = pb[n * 3 + 0], py = pb[n * 3 + 1], pz = pb[n * 3 + 2];
  __syncthreads();

  // --- phase B: gather k/v rows (float4); qk = q - kj ; vv = vj
  {
    int j = lane >> 1, c0 = (lane & 1) * 32;
    int id = s_id[w][j];
    const float* krow = qkv + ((size_t)(b * NPTS + id)) * 192 + 64;
    const float* vrow = krow + 64;
#pragma unroll
    for (int i = 0; i < 8; ++i) {
      int c = c0 + i * 4;
      float4 k4 = *(const float4*)(krow + c);
      float4 v4 = *(const float4*)(vrow + c);
      float4 q4 = *(const float4*)(&s_q[w][c]);
      float4 qk = { q4.x - k4.x, q4.y - k4.y, q4.z - k4.z, q4.w - k4.w };
      *(float4*)(&s_qk[w][j][c]) = qk;
      *(float4*)(&s_vv[w][j][c]) = v4;
    }
    if (c0 == 0) {
      s_rel[w][j][0] = px - pb[id * 3 + 0];
      s_rel[w][j][1] = py - pb[id * 3 + 1];
      s_rel[w][j][2] = pz - pb[id * 3 + 2];
    }
  }
  __syncthreads();

  // --- phase C: pe hidden = relu(rel@pos_w1 + pos_b1)  (16x32, VALU)
  {
    int j = lane >> 1, h0 = (lane & 1) * 16;
    float rx = s_rel[w][j][0], ry = s_rel[w][j][1], rz = s_rel[w][j][2];
#pragma unroll
    for (int hh = 0; hh < 16; ++hh) {
      int h = h0 + hh;
      float v = rx * pos_w1[h] + ry * pos_w1[32 + h] + rz * pos_w1[64 + h] + pos_b1[h];
      s_peh[w][j][h] = (_Float16)fmaxf(v, 0.0f);
    }
  }
  __syncthreads();

  // --- phase D: pe = peh @ pos_w2 + pos_b2 (WMMA K=32); fold into vv and x1
  {
    v16h aP = load_a_frag(&s_peh[w][0][0], 32, 0);
    for (int t = 0; t < 4; ++t) {
      v8f d = {};
      d = wmma16(aP, load_bt_frag(pos_w2t, 32, t * 16, 0), d);
      int c = t * 16 + nn;
      float b2 = pos_b2[c];
#pragma unroll
      for (int r = 0; r < 8; ++r) {
        int j = r + 8 * g;
        float pe = d[r] + b2;
        s_vv[w][j][c] += pe;
        s_x1[w][j][c] = (_Float16)(s_qk[w][j][c] + pe);
      }
    }
  }
  __syncthreads();

  // --- phase E: hid = relu(x1 @ attn_w1 + b1)  16x256, K=64
  {
    v16h a0 = load_a_frag(&s_x1[w][0][0], 64, 0);
    v16h a1 = load_a_frag(&s_x1[w][0][0], 64, 32);
    for (int t = 0; t < 16; ++t) {
      v8f d = {};
      d = wmma16(a0, load_bt_frag(w1t, 64, t * 16, 0), d);
      d = wmma16(a1, load_bt_frag(w1t, 64, t * 16, 32), d);
      int c = t * 16 + nn;
      float bb = attn_b1[c];
#pragma unroll
      for (int r = 0; r < 8; ++r)
        s_hid[w][r + 8 * g][c] = (_Float16)fmaxf(d[r] + bb, 0.0f);
    }
  }
  __syncthreads();

  // --- phase F: sim = hid @ attn_w2 + b2  16x64, K=256
  {
    v16h aH[8];
#pragma unroll
    for (int kk = 0; kk < 8; ++kk)
      aH[kk] = load_a_frag(&s_hid[w][0][0], 256, kk * 32);
    for (int t = 0; t < 4; ++t) {
      v8f d = {};
#pragma unroll
      for (int kk = 0; kk < 8; ++kk)
        d = wmma16(aH[kk], load_bt_frag(w2t, 256, t * 16, kk * 32), d);
      int c = t * 16 + nn;
      float bb = attn_b2[c];
#pragma unroll
      for (int r = 0; r < 8; ++r)
        s_sim[w][r + 8 * g][c] = d[r] + bb;
    }
  }
  __syncthreads();

  // --- phase G: per-channel softmax over the 16 neighbors, aggregate
  for (int cc = 0; cc < 2; ++cc) {
    int c = lane + cc * 32;
    float m = -3.0e38f;
#pragma unroll
    for (int j = 0; j < 16; ++j) m = fmaxf(m, s_sim[w][j][c]);
    float s = 0.f, a = 0.f;
#pragma unroll
    for (int j = 0; j < 16; ++j) {
      float e = __expf(s_sim[w][j][c] - m);
      s += e;
      a += e * s_vv[w][j][c];
    }
    agg[(size_t)pt * 64 + c] = a / s;
  }
}

// ---------------------------------------------------------------------------
// kernel_launch
// ---------------------------------------------------------------------------
extern "C" void kernel_launch(void* const* d_in, const int* in_sizes, int n_in,
                              void* d_out, int out_size, void* d_ws, size_t ws_size,
                              hipStream_t stream) {
  const float* x       = (const float*)d_in[0];
  const float* p       = (const float*)d_in[1];
  const float* fc1_w   = (const float*)d_in[2];
  const float* fc1_b   = (const float*)d_in[3];
  const float* bn1_g   = (const float*)d_in[4];
  const float* bn1_b   = (const float*)d_in[5];
  const float* qkv_w   = (const float*)d_in[6];
  const float* pos_w1  = (const float*)d_in[7];
  const float* pos_b1  = (const float*)d_in[8];
  const float* pos_w2  = (const float*)d_in[9];
  const float* pos_b2  = (const float*)d_in[10];
  const float* attn_w1 = (const float*)d_in[11];
  const float* attn_b1 = (const float*)d_in[12];
  const float* attn_w2 = (const float*)d_in[13];
  const float* attn_b2 = (const float*)d_in[14];
  const float* bn2_g   = (const float*)d_in[15];
  const float* bn2_b   = (const float*)d_in[16];
  const float* fc2_w   = (const float*)d_in[17];
  const float* fc2_b   = (const float*)d_in[18];
  const float* bn3_g   = (const float*)d_in[19];
  const float* bn3_b   = (const float*)d_in[20];

  const int R = R_TOTAL;
  char* ws = (char*)d_ws;
  size_t off = 0;
  auto alloc = [&](size_t bytes) -> char* {
    char* q = ws + off;
    off = (off + bytes + 255) & ~(size_t)255;
    return q;
  };

  _Float16* xh     = (_Float16*)alloc((size_t)R * 64 * 2);
  float*    hpre   = (float*)   alloc((size_t)R * 64 * 4);
  float*    hf32   = (float*)   alloc((size_t)R * 64 * 4);
  _Float16* hf16   = (_Float16*)alloc((size_t)R * 64 * 2);
  float*    qkvb   = (float*)   alloc((size_t)R * 192 * 4);
  int*      idxb   = (int*)     alloc((size_t)R * KNN * 4);
  float*    aggb   = (float*)   alloc((size_t)R * 64 * 4);
  _Float16* sf16   = (_Float16*)alloc((size_t)R * 64 * 2);
  float*    ypre   = (float*)   alloc((size_t)R * 64 * 4);
  _Float16* fc1wt  = (_Float16*)alloc(64 * 64 * 2);
  _Float16* qkvwt  = (_Float16*)alloc(192 * 64 * 2);
  _Float16* fc2wt  = (_Float16*)alloc(64 * 64 * 2);
  _Float16* posw2t = (_Float16*)alloc(64 * 32 * 2);
  _Float16* w1t    = (_Float16*)alloc(256 * 64 * 2);
  _Float16* w2t    = (_Float16*)alloc(64 * 256 * 2);
  float*    stats  = (float*)   alloc((6 * 64 + 192) * 4);
  float *s1s = stats, *s1q = stats + 64;
  float *s2s = stats + 128, *s2q = stats + 192;
  float *s3s = stats + 256, *s3q = stats + 320;
  float *zbias = stats + 384;   // 192 zeros: branch-free "no bias"

  // 0. zero stats + dummy bias
  zero_f32<<<3, 256, 0, stream>>>(stats, 6 * 64 + 192);

  // 1. weight prep (f16, transposed)
  wt_prep<<<16, 256, 0, stream>>>(fc1_w, fc1wt, 64, 64);
  wt_prep<<<48, 256, 0, stream>>>(qkv_w, qkvwt, 64, 192);
  wt_prep<<<16, 256, 0, stream>>>(fc2_w, fc2wt, 64, 64);
  wt_prep<<<8, 256, 0, stream>>>(pos_w2, posw2t, 32, 64);
  wt_prep<<<64, 256, 0, stream>>>(attn_w1, w1t, 64, 256);
  wt_prep<<<64, 256, 0, stream>>>(attn_w2, w2t, 256, 64);

  // 2. x -> f16
  f32_to_f16_v4<<<(R * 64 / 4) / 256, 256, 0, stream>>>(x, xh, R * 64 / 4);

  // 3. fc1: hpre = x @ fc1_w + fc1_b  (WMMA)
  gemm_k64<<<R / 128, 256, 0, stream>>>(xh, fc1wt, fc1_b, hpre, R, 64);

  // 4. bn1 stats + apply (h = relu(bn1(hpre)))
  chan_stats<<<64, 256, 0, stream>>>(hpre, s1s, s1q, R);
  bn_apply_v4<<<(R * 64 / 4) / 256, 256, 0, stream>>>(hpre, s1s, s1q, bn1_g, bn1_b,
                                                      nullptr, hf32, hf16, R);

  // 5. qkv = h @ qkv_w  (WMMA, no bias)
  gemm_k64<<<R / 128, 256, 0, stream>>>(hf16, qkvwt, zbias, qkvb, R, 192);

  // 6. knn
  knn_kernel<<<R / 256, 256, 0, stream>>>(p, idxb);

  // 7. fused point-transformer attention (WMMA)
  attn_kernel<<<R / AW, 64, 0, stream>>>(qkvb, p, idxb, pos_w1, pos_b1, posw2t,
                                         pos_b2, w1t, attn_b1, w2t, attn_b2, aggb);

  // 8. bn2 + residual: s = relu(bn2(agg)) + h
  chan_stats<<<64, 256, 0, stream>>>(aggb, s2s, s2q, R);
  bn_apply_v4<<<(R * 64 / 4) / 256, 256, 0, stream>>>(aggb, s2s, s2q, bn2_g, bn2_b,
                                                      hf32, nullptr, sf16, R);

  // 9. fc2: ypre = s @ fc2_w + fc2_b  (WMMA)
  gemm_k64<<<R / 128, 256, 0, stream>>>(sf16, fc2wt, fc2_b, ypre, R, 64);

  // 10. bn3: y = relu(bn3(ypre)) -> d_out
  chan_stats<<<64, 256, 0, stream>>>(ypre, s3s, s3q, R);
  bn_apply_v4<<<(R * 64 / 4) / 256, 256, 0, stream>>>(ypre, s3s, s3q, bn3_g, bn3_b,
                                                      nullptr, (float*)d_out, nullptr, R);

  // 11. second output: p passthrough
  copy_f32_v4<<<(R * 3 / 4) / 256, 256, 0, stream>>>(p, (float*)d_out + (size_t)R * 64,
                                                     R * 3 / 4);
}
